// FreqAttention_2216203125335
// MI455X (gfx1250) — matrix operand order repaired and checked by reading, not compiled, not run on hardware
//
#include <hip/hip_runtime.h>
#include <hip/hip_bf16.h>
#include <math.h>
#include <stdint.h>

// ---------------------------------------------------------------------------
// FreqAttention on MI455X (gfx1250, wave32).
// Entire pipeline mapped onto v_wmma_f32_16x16x32_f16:
//   LN -> DFT-as-WMMA (twiddles generated in registers) -> 6 GEMMs ->
//   flash-softmax attention (TDM-staged K/V tiles, WMMA scores + WMMA P@V) ->
//   iDFT-as-WMMA -> final GEMM (f32 out).
// ---------------------------------------------------------------------------

typedef __attribute__((ext_vector_type(16))) _Float16 v16h;
typedef __attribute__((ext_vector_type(8)))  float    v8f;
typedef uint32_t u32x4 __attribute__((ext_vector_type(4)));
typedef uint32_t u32x8 __attribute__((ext_vector_type(8)));

#define WMMA(A, B, C) __builtin_amdgcn_wmma_f32_16x16x32_f16( \
    false, (A), false, (B), (short)0, (C), false, false)

static constexpr int B_  = 4;
static constexpr int L_  = 4096;
static constexpr int E_  = 512;
static constexpr int H_  = 8;
static constexpr int D_  = 64;
static constexpr int FR_ = L_ / 2 + 1;     // 2049 real frequency bins
static constexpr int FP_ = 2080;           // padded to 65*32
static constexpr float TWO_PI = 6.28318530717958647692f;

// A-fragment k index for 16-bit 16x32 A matrix (lane half = lane>>4)
__device__ __forceinline__ int a_kmap(int h, int half) {
  return half * 16 + h;
}

// ---------------------------------------------------------------------------
// Tensor Data Mover: async-load a 32-row x 64-col f16 tile (row stride 512
// elements) from global into LDS.  D# layout per CDNA5 ISA §8.3/8.4.
// 2D tile -> VADDR2/VADDR3 omitted (NULL).  Tracked by TENSORcnt.
// ---------------------------------------------------------------------------
__device__ __forceinline__ void tdm_load_tile_32x64(uint32_t lds_addr,
                                                    const _Float16* gsrc) {
  const uint64_t ga = (uint64_t)(uintptr_t)gsrc;
  u32x4 g0;
  g0[0] = 1u;                                    // count=1, user descriptor
  g0[1] = lds_addr;                              // D#.lds_addr
  g0[2] = (uint32_t)ga;                          // global_addr[31:0]
  g0[3] = (uint32_t)((ga >> 32) & 0x01FFFFFFu)   // global_addr[56:32]
        | (2u << 30);                            // type = 2 ("image")
  u32x8 g1;
  g1[0] = 1u << 16;                              // wg_mask=0, data_size=1 (2B)
  g1[1] = (uint32_t)D_ << 16;                    // tensor_dim0[15:0] = 64
  g1[2] = 32u << 16;                             // dim0 hi=0, tensor_dim1 = 32
  g1[3] = (uint32_t)D_ << 16;                    // dim1 hi=0, tile_dim0 = 64
  g1[4] = 32u;                                   // tile_dim1=32, tile_dim2=0
  g1[5] = (uint32_t)E_;                          // tensor_dim0_stride = 512
  g1[6] = 0u;
  g1[7] = 0u;
  asm volatile("tensor_load_to_lds %0, %1" :: "s"(g0), "s"(g1) : "memory");
}

// ---------------------------------------------------------------------------
// 1. LayerNorm (f32 in) -> f16 out.  One 256-thread block per row of 512.
// ---------------------------------------------------------------------------
__global__ void ln_kernel(const float* __restrict__ x,
                          const float* __restrict__ gamma,
                          const float* __restrict__ beta,
                          _Float16* __restrict__ xn) {
  const int row = blockIdx.x;
  const int tid = threadIdx.x;
  const float* xr = x + (size_t)row * E_;
  float a = xr[tid], b = xr[tid + 256];
  __shared__ float red[256];
  red[tid] = a + b;
  __syncthreads();
  for (int off = 128; off > 0; off >>= 1) {
    if (tid < off) red[tid] += red[tid + off];
    __syncthreads();
  }
  const float mu = red[0] * (1.0f / E_);
  __syncthreads();
  const float da = a - mu, db = b - mu;
  red[tid] = da * da + db * db;
  __syncthreads();
  for (int off = 128; off > 0; off >>= 1) {
    if (tid < off) red[tid] += red[tid + off];
    __syncthreads();
  }
  const float inv = rsqrtf(red[0] * (1.0f / E_) + 1e-5f);
  _Float16* o = xn + (size_t)row * E_;
  o[tid]       = (_Float16)(da * inv * gamma[tid] + beta[tid]);
  o[tid + 256] = (_Float16)(db * inv * gamma[tid + 256] + beta[tid + 256]);
}

// ---------------------------------------------------------------------------
// 2. f32 -> f16 weight conversion
// ---------------------------------------------------------------------------
__global__ void cvt_kernel(const float* __restrict__ src,
                           _Float16* __restrict__ dst, int n) {
  int i = blockIdx.x * blockDim.x + threadIdx.x;
  if (i < n) dst[i] = (_Float16)src[i];
}

// ---------------------------------------------------------------------------
// 3. Forward real DFT as WMMA:  xf[b,f,e] = sum_l e^{-2pi i f l/L} xn[b,l,e]/sqrt(L)
//    Twiddle A-fragments generated in registers; phase reduced mod L exactly.
//    grid = (E/16, FP/16, B), block = 32 (one wave per 16x16 output tile).
// ---------------------------------------------------------------------------
__global__ void rdft_kernel(const _Float16* __restrict__ xn,
                            _Float16* __restrict__ xr,
                            _Float16* __restrict__ xi) {
  const int et = blockIdx.x, ft = blockIdx.y, b = blockIdx.z;
  const int lane = threadIdx.x, hf = lane >> 4, l15 = lane & 15;
  const int f    = ft * 16 + l15;   // A-row this lane owns
  const int ecol = et * 16 + l15;   // B/C column this lane owns
  const _Float16* xb = xn + (size_t)b * L_ * E_;
  v8f cr = {}, ci = {};
  for (int kb = 0; kb < L_; kb += 32) {
    v16h ac, asn, bf;
#pragma unroll
    for (int h = 0; h < 16; ++h) {
      const int k = kb + a_kmap(h, hf);
      const int p = (f * k) & (L_ - 1);
      const float ang = -TWO_PI * (float)p * (1.0f / L_);
      ac[h]  = (_Float16)__cosf(ang);
      asn[h] = (_Float16)__sinf(ang);
    }
#pragma unroll
    for (int h = 0; h < 16; ++h) {
      const int k = kb + 16 * hf + h;
      bf[h] = xb[(size_t)k * E_ + ecol];
    }
    cr = WMMA(ac, bf, cr);
    ci = WMMA(asn, bf, ci);
  }
  const size_t base = (size_t)b * FP_ * E_;
  const float sc = 1.0f / 64.0f;    // ortho norm 1/sqrt(4096)
#pragma unroll
  for (int v = 0; v < 8; ++v) {
    const int fr = ft * 16 + v + 8 * hf;
    float vr = cr[v] * sc, vi = ci[v] * sc;
    if (fr >= FR_) { vr = 0.f; vi = 0.f; }   // zero the padded bins
    xr[base + (size_t)fr * E_ + ecol] = (_Float16)vr;
    xi[base + (size_t)fr * E_ + ecol] = (_Float16)vi;
  }
}

// ---------------------------------------------------------------------------
// 4. GEMM  Y[r,o] = sum_e X[r,e] * W[o,e] + bias[o]   (torch Linear)
//    f16 in, f32 accum; f16 out (projections) or f32 out (final Wo).
//    grid = (32, rowtiles), block = 32.
// ---------------------------------------------------------------------------
__global__ void gemm_kernel(const _Float16* __restrict__ X,
                            const _Float16* __restrict__ W,
                            const float* __restrict__ bias,
                            _Float16* __restrict__ Yh,
                            float* __restrict__ Yf) {
  const int ot = blockIdx.x, rt = blockIdx.y;
  const int lane = threadIdx.x, hf = lane >> 4, l15 = lane & 15;
  const int ocol = ot * 16 + l15;
  const int arow = rt * 16 + l15;
  v8f c;
  const float bv = bias ? bias[ocol] : 0.0f;
#pragma unroll
  for (int v = 0; v < 8; ++v) c[v] = bv;
  const _Float16* Xr = X + (size_t)arow * E_;
  const _Float16* Wr = W + (size_t)ocol * E_;
  for (int kb = 0; kb < E_; kb += 32) {
    __builtin_prefetch(Xr + kb + 64, 0, 1);   // global_prefetch_b8
    v16h af, bf;
#pragma unroll
    for (int h = 0; h < 16; ++h) af[h] = Xr[kb + a_kmap(h, hf)];
#pragma unroll
    for (int h = 0; h < 16; ++h) bf[h] = Wr[kb + 16 * hf + h];
    c = WMMA(af, bf, c);
  }
#pragma unroll
  for (int v = 0; v < 8; ++v) {
    const int row = rt * 16 + v + 8 * hf;
    if (Yh) Yh[(size_t)row * E_ + ocol] = (_Float16)c[v];
    else    Yf[(size_t)row * E_ + ocol] = c[v];
  }
}

// ---------------------------------------------------------------------------
// 5. Flash attention over frequency bins. One wave per (16-query tile, h, b).
//    K/V chunk tiles (32 x 64 f16) staged Global->LDS by the Tensor Data
//    Mover each iteration; scores = Qr.Kr + Qi.Ki (scale folded into Q),
//    online softmax, P staged C-layout -> A-layout through LDS, then P@V.
// ---------------------------------------------------------------------------
__global__ void attn_kernel(const _Float16* __restrict__ Qr,
                            const _Float16* __restrict__ Qi,
                            const _Float16* __restrict__ Kr,
                            const _Float16* __restrict__ Ki,
                            const _Float16* __restrict__ Vr,
                            const _Float16* __restrict__ Vi,
                            _Float16* __restrict__ Or,
                            _Float16* __restrict__ Oi) {
  const int lt = blockIdx.x, h = blockIdx.y, b = blockIdx.z;
  const int lane = threadIdx.x, hf = lane >> 4, l15 = lane & 15;
  const size_t rowbase = (size_t)b * FP_;
  const int col0 = h * D_;

  __shared__ _Float16 KrL[32 * D_];
  __shared__ _Float16 KiL[32 * D_];
  __shared__ _Float16 VrL[32 * D_];
  __shared__ _Float16 ViL[32 * D_];
  __shared__ _Float16 Plds[16 * 32];
  const uint32_t krA = (uint32_t)(uintptr_t)&KrL[0];
  const uint32_t kiA = (uint32_t)(uintptr_t)&KiL[0];
  const uint32_t vrA = (uint32_t)(uintptr_t)&VrL[0];
  const uint32_t viA = (uint32_t)(uintptr_t)&ViL[0];

  // Q fragments, scale d^-1/2 = 1/8 folded in (exact in f16)
  v16h aqr[2], aqi[2];
#pragma unroll
  for (int dc = 0; dc < 2; ++dc) {
#pragma unroll
    for (int hh = 0; hh < 16; ++hh) {
      const int k = dc * 32 + a_kmap(hh, hf);
      const size_t ro = (rowbase + lt * 16 + l15) * E_ + col0 + k;
      aqr[dc][hh] = (_Float16)((float)Qr[ro] * 0.125f);
      aqi[dc][hh] = (_Float16)((float)Qi[ro] * 0.125f);
    }
  }

  v8f o_r[4] = {}, o_i[4] = {};
  float m_run[8], s_run[8];
#pragma unroll
  for (int v = 0; v < 8; ++v) { m_run[v] = -3.0e38f; s_run[v] = 0.0f; }

  for (int mc = 0; mc < FP_ / 32; ++mc) {          // 65 chunks of 32 keys
    // drain previous iteration's LDS reads, then DMA this chunk's tiles
    __syncthreads();
    const size_t gro = (rowbase + mc * 32) * E_ + col0;
    tdm_load_tile_32x64(krA, Kr + gro);
    tdm_load_tile_32x64(kiA, Ki + gro);
    tdm_load_tile_32x64(vrA, Vr + gro);
    tdm_load_tile_32x64(viA, Vi + gro);
    __builtin_amdgcn_s_wait_tensorcnt(0);

    v8f s0 = {}, s1 = {};
#pragma unroll
    for (int t = 0; t < 2; ++t) {
#pragma unroll
      for (int dc = 0; dc < 2; ++dc) {
        v16h bkr, bki;
#pragma unroll
        for (int hh = 0; hh < 16; ++hh) {
          const int k = dc * 32 + 16 * hf + hh;        // d index
          const int m = t * 16 + l15;                  // key row in tile
          bkr[hh] = KrL[m * D_ + k];
          bki[hh] = KiL[m * D_ + k];
        }
        if (t == 0) { s0 = WMMA(aqr[dc], bkr, s0); s0 = WMMA(aqi[dc], bki, s0); }
        else        { s1 = WMMA(aqr[dc], bkr, s1); s1 = WMMA(aqi[dc], bki, s1); }
      }
    }
    // mask padded keys (column index = lane&15 in C layout)
    const int m0 = mc * 32 + l15, m1 = mc * 32 + 16 + l15;
    if (m0 >= FR_) {
#pragma unroll
      for (int v = 0; v < 8; ++v) s0[v] = -3.0e38f;
    }
    if (m1 >= FR_) {
#pragma unroll
      for (int v = 0; v < 8; ++v) s1[v] = -3.0e38f;
    }
    // online softmax: row stats live across a 16-lane half per VGPR index
    float alpha[8];
#pragma unroll
    for (int v = 0; v < 8; ++v) {
      float mx = fmaxf(s0[v], s1[v]);
#pragma unroll
      for (int msk = 1; msk < 16; msk <<= 1)
        mx = fmaxf(mx, __shfl_xor(mx, msk, 32));
      const float mnew = fmaxf(m_run[v], mx);
      alpha[v] = __expf(m_run[v] - mnew);
      const float p0 = __expf(s0[v] - mnew);
      const float p1 = __expf(s1[v] - mnew);
      float ps = p0 + p1;
#pragma unroll
      for (int msk = 1; msk < 16; msk <<= 1)
        ps += __shfl_xor(ps, msk, 32);
      s_run[v] = s_run[v] * alpha[v] + ps;
      m_run[v] = mnew;
      Plds[(v + 8 * hf) * 32 + l15]      = (_Float16)p0;
      Plds[(v + 8 * hf) * 32 + 16 + l15] = (_Float16)p1;
#pragma unroll
      for (int nt = 0; nt < 4; ++nt) {
        o_r[nt][v] *= alpha[v];
        o_i[nt][v] *= alpha[v];
      }
    }
    __syncthreads();
    v16h ap;
#pragma unroll
    for (int hh = 0; hh < 16; ++hh)
      ap[hh] = Plds[l15 * 32 + a_kmap(hh, hf)];
#pragma unroll
    for (int nt = 0; nt < 4; ++nt) {
      v16h bvr, bvi;
#pragma unroll
      for (int hh = 0; hh < 16; ++hh) {
        const int k = 16 * hf + hh;                    // key row in tile
        const int n = nt * 16 + l15;                   // d column
        bvr[hh] = VrL[k * D_ + n];
        bvi[hh] = ViL[k * D_ + n];
      }
      o_r[nt] = WMMA(ap, bvr, o_r[nt]);
      o_i[nt] = WMMA(ap, bvi, o_i[nt]);
    }
  }
  // normalize and store (padded rows untouched; buffer pre-zeroed by rdft)
#pragma unroll
  for (int v = 0; v < 8; ++v) {
    const int fr = lt * 16 + v + 8 * hf;
    if (fr >= FR_) continue;
    const float inv = 1.0f / s_run[v];
    const size_t ro = (rowbase + fr) * E_ + col0;
#pragma unroll
    for (int nt = 0; nt < 4; ++nt) {
      Or[ro + nt * 16 + l15] = (_Float16)(o_r[nt][v] * inv);
      Oi[ro + nt * 16 + l15] = (_Float16)(o_i[nt][v] * inv);
    }
  }
}

// ---------------------------------------------------------------------------
// 6. Inverse real DFT as WMMA:
//    x[t] = (1/sqrt(L)) sum_f w_f (cos(2pi f t/L) Xr[f] - sin(2pi f t/L) Xi[f])
//    w_f = 1 for f in {0, L/2}, 2 otherwise, 0 for padded bins.
// ---------------------------------------------------------------------------
__global__ void irdft_kernel(const _Float16* __restrict__ Or,
                             const _Float16* __restrict__ Oi,
                             _Float16* __restrict__ outt) {
  const int et = blockIdx.x, tt = blockIdx.y, b = blockIdx.z;
  const int lane = threadIdx.x, hf = lane >> 4, l15 = lane & 15;
  const int t    = tt * 16 + l15;
  const int ecol = et * 16 + l15;
  const size_t base = (size_t)b * FP_ * E_;
  v8f c = {};
  for (int fb = 0; fb < FP_; fb += 32) {
    v16h ac, asn, br, bi;
#pragma unroll
    for (int hh = 0; hh < 16; ++hh) {
      const int f = fb + a_kmap(hh, hf);
      float wf = (f == 0 || f == L_ / 2) ? 1.0f : 2.0f;
      if (f >= FR_) wf = 0.0f;
      const int p = (f * t) & (L_ - 1);
      const float ang = TWO_PI * (float)p * (1.0f / L_);
      ac[hh]  = (_Float16)(wf * __cosf(ang));
      asn[hh] = (_Float16)(-wf * __sinf(ang));
    }
#pragma unroll
    for (int hh = 0; hh < 16; ++hh) {
      const int f = fb + 16 * hf + hh;
      br[hh] = Or[base + (size_t)f * E_ + ecol];
      bi[hh] = Oi[base + (size_t)f * E_ + ecol];
    }
    c = WMMA(ac, br, c);
    c = WMMA(asn, bi, c);
  }
#pragma unroll
  for (int v = 0; v < 8; ++v) {
    const int tg = tt * 16 + v + 8 * hf;
    outt[((size_t)b * L_ + tg) * E_ + ecol] = (_Float16)(c[v] * (1.0f / 64.0f));
  }
}

// ---------------------------------------------------------------------------
// Host-side orchestration
// ---------------------------------------------------------------------------
extern "C" void kernel_launch(void* const* d_in, const int* in_sizes, int n_in,
                              void* d_out, int out_size, void* d_ws, size_t ws_size,
                              hipStream_t stream) {
  (void)in_sizes; (void)n_in; (void)out_size; (void)ws_size;
  const float* x   = (const float*)d_in[0];
  const float* lg  = (const float*)d_in[1];
  const float* lb  = (const float*)d_in[2];
  const float* Wqr = (const float*)d_in[3];  const float* bqr = (const float*)d_in[4];
  const float* Wqi = (const float*)d_in[5];  const float* bqi = (const float*)d_in[6];
  const float* Wkr = (const float*)d_in[7];  const float* bkr = (const float*)d_in[8];
  const float* Wki = (const float*)d_in[9];  const float* bki = (const float*)d_in[10];
  const float* Wvr = (const float*)d_in[11]; const float* bvr = (const float*)d_in[12];
  const float* Wvi = (const float*)d_in[13]; const float* bvi = (const float*)d_in[14];
  const float* Wo  = (const float*)d_in[15];
  float* out = (float*)d_out;

  char* ws = (char*)d_ws;
  const size_t XN_B  = (size_t)B_ * L_ * E_ * 2;    // 16.78 MB, reused as out_t
  const size_t FBUF  = (size_t)B_ * FP_ * E_ * 2;   // 8.52 MB each
  _Float16* xn  = (_Float16*)(ws);
  _Float16* xr  = (_Float16*)(ws + XN_B);
  _Float16* xi  = (_Float16*)(ws + XN_B + FBUF);
  _Float16* qr  = (_Float16*)(ws + XN_B + 2 * FBUF);
  _Float16* qi  = (_Float16*)(ws + XN_B + 3 * FBUF);
  _Float16* kr  = (_Float16*)(ws + XN_B + 4 * FBUF);
  _Float16* ki  = (_Float16*)(ws + XN_B + 5 * FBUF);
  _Float16* vr  = (_Float16*)(ws + XN_B + 6 * FBUF);
  _Float16* vi  = (_Float16*)(ws + XN_B + 7 * FBUF);
  _Float16* w16 = (_Float16*)(ws + XN_B + 8 * FBUF);
  const int WSZ = E_ * E_;
  _Float16* w_qr = w16 + 0 * WSZ;
  _Float16* w_qi = w16 + 1 * WSZ;
  _Float16* w_kr = w16 + 2 * WSZ;
  _Float16* w_ki = w16 + 3 * WSZ;
  _Float16* w_vr = w16 + 4 * WSZ;
  _Float16* w_vi = w16 + 5 * WSZ;
  _Float16* w_o  = w16 + 6 * WSZ;
  _Float16* out_t = xn;           // xn dead after rdft; reuse for time-domain out
  _Float16* o_r   = xr;           // xr/xi dead after projections; reuse for attn out
  _Float16* o_i   = xi;

  // 1. LayerNorm
  ln_kernel<<<B_ * L_, 256, 0, stream>>>(x, lg, lb, xn);

  // 2. Weight conversion
  const int cb = (WSZ + 255) / 256;
  cvt_kernel<<<cb, 256, 0, stream>>>(Wqr, w_qr, WSZ);
  cvt_kernel<<<cb, 256, 0, stream>>>(Wqi, w_qi, WSZ);
  cvt_kernel<<<cb, 256, 0, stream>>>(Wkr, w_kr, WSZ);
  cvt_kernel<<<cb, 256, 0, stream>>>(Wki, w_ki, WSZ);
  cvt_kernel<<<cb, 256, 0, stream>>>(Wvr, w_vr, WSZ);
  cvt_kernel<<<cb, 256, 0, stream>>>(Wvi, w_vi, WSZ);
  cvt_kernel<<<cb, 256, 0, stream>>>(Wo,  w_o,  WSZ);

  // 3. Forward DFT (WMMA with register-generated twiddles)
  rdft_kernel<<<dim3(E_ / 16, FP_ / 16, B_), 32, 0, stream>>>(xn, xr, xi);

  // 4. Six QKV projections (WMMA GEMM, f16 out)
  const dim3 pg(E_ / 16, (B_ * FP_) / 16);
  gemm_kernel<<<pg, 32, 0, stream>>>(xr, w_qr, bqr, qr, nullptr);
  gemm_kernel<<<pg, 32, 0, stream>>>(xi, w_qi, bqi, qi, nullptr);
  gemm_kernel<<<pg, 32, 0, stream>>>(xr, w_kr, bkr, kr, nullptr);
  gemm_kernel<<<pg, 32, 0, stream>>>(xi, w_ki, bki, ki, nullptr);
  gemm_kernel<<<pg, 32, 0, stream>>>(xr, w_vr, bvr, vr, nullptr);
  gemm_kernel<<<pg, 32, 0, stream>>>(xi, w_vi, bvi, vi, nullptr);

  // 5. Frequency-domain attention (flash softmax, TDM-staged K/V, all-WMMA)
  attn_kernel<<<dim3(FP_ / 16, H_, B_), 32, 0, stream>>>(
      qr, qi, kr, ki, vr, vi, o_r, o_i);

  // 6. Inverse DFT (WMMA)
  irdft_kernel<<<dim3(E_ / 16, L_ / 16, B_), 32, 0, stream>>>(o_r, o_i, out_t);

  // 7. Final projection Wo (f32 out)
  gemm_kernel<<<dim3(E_ / 16, (B_ * L_) / 16), 32, 0, stream>>>(
      out_t, w_o, nullptr, nullptr, out);
}
